// GraphSAGE_48009144434901
// MI455X (gfx1250) — matrix operand order repaired and checked by reading, not compiled, not run on hardware
//
#include <hip/hip_runtime.h>
#include <hip/hip_bf16.h>

// ---------------------------------------------------------------------------
// GraphSAGE (2-layer, mean aggregation) for MI455X / gfx1250, wave32 + WMMA.
//   N=100000 nodes, E=1600000 edges, F_IN=128, HID=256, C_OUT=4
// GEMMs run as v_wmma_f32_16x16x32_f16 (f16 in, f32 accumulate).
// Edge scatter runs as f32 global atomics; h1 kept in f16 so the whole
// feature working set (h1 51MB + agg 102MB) stays L2-resident (192MB L2).
// ---------------------------------------------------------------------------

#define NN 100000
#define EE 1600000
#define F_IN 128
#define HID 256
#define MTILES (NN / 16)   // 6250, exact

typedef __attribute__((ext_vector_type(16))) _Float16 v16h;
typedef __attribute__((ext_vector_type(8)))  _Float16 v8h;
typedef __attribute__((ext_vector_type(8)))  float    v8f;

// ---- WMMA A-fragment builders (ISA 7.12.2, wave32 16-bit A 16x32 layout) ---
// Lane L holds row m = L%16.  With h = L/16:
//   packed elements i=0..7  -> k = kbase + 8*h + i
//   packed elements i=8..15 -> k = kbase + 8*h + 8 + i   (i.e. +16 region)
__device__ inline v16h load_a_f32(const float* __restrict__ row, int k0, float s) {
    float4 f0 = *(const float4*)(row + k0);
    float4 f1 = *(const float4*)(row + k0 + 4);
    float4 f2 = *(const float4*)(row + k0 + 16);
    float4 f3 = *(const float4*)(row + k0 + 20);
    v16h a;
    a[0]=(_Float16)(f0.x*s);  a[1]=(_Float16)(f0.y*s);
    a[2]=(_Float16)(f0.z*s);  a[3]=(_Float16)(f0.w*s);
    a[4]=(_Float16)(f1.x*s);  a[5]=(_Float16)(f1.y*s);
    a[6]=(_Float16)(f1.z*s);  a[7]=(_Float16)(f1.w*s);
    a[8]=(_Float16)(f2.x*s);  a[9]=(_Float16)(f2.y*s);
    a[10]=(_Float16)(f2.z*s); a[11]=(_Float16)(f2.w*s);
    a[12]=(_Float16)(f3.x*s); a[13]=(_Float16)(f3.y*s);
    a[14]=(_Float16)(f3.z*s); a[15]=(_Float16)(f3.w*s);
    return a;
}

__device__ inline v16h load_a_f16(const _Float16* __restrict__ row, int k0) {
    v8h a = *(const v8h*)(row + k0);
    v8h b = *(const v8h*)(row + k0 + 16);
    return __builtin_shufflevector(a, b, 0,1,2,3,4,5,6,7,8,9,10,11,12,13,14,15);
}

// ------------------------------- utility -----------------------------------
__global__ void zero_f4_kernel(float4* __restrict__ p, int n4) {
    int i = blockIdx.x * blockDim.x + threadIdx.x;
    if (i < n4) p[i] = make_float4(0.f, 0.f, 0.f, 0.f);
}

__global__ void deg_kernel(const long long* __restrict__ ei, float* __restrict__ deg) {
    int e = blockIdx.x * blockDim.x + threadIdx.x;
    if (e < EE) atomicAdd(&deg[(int)ei[EE + e]], 1.0f);
}

__global__ void invdeg_kernel(float* __restrict__ deg) {
    int i = blockIdx.x * blockDim.x + threadIdx.x;
    if (i < NN) { float d = deg[i]; deg[i] = d > 0.f ? 1.f / d : 0.f; }
}

// Convert weights to f16 (W2 zero-padded from 4 to 16 output columns).
__global__ void prep_weights_kernel(const float* __restrict__ w1l, const float* __restrict__ w1r,
                                    const float* __restrict__ w2l, const float* __restrict__ w2r,
                                    _Float16* __restrict__ W1L, _Float16* __restrict__ W1R,
                                    _Float16* __restrict__ W2L, _Float16* __restrict__ W2R) {
    int i = blockIdx.x * blockDim.x + threadIdx.x;
    if (i < HID * F_IN) {
        W1L[i] = (_Float16)w1l[i];
        W1R[i] = (_Float16)w1r[i];
    }
    if (i < 16 * HID) {
        int n = i >> 8;           // padded output column 0..15
        int k = i & (HID - 1);
        W2L[i] = (_Float16)(n < 4 ? w2l[n * HID + k] : 0.f);
        W2R[i] = (_Float16)(n < 4 ? w2r[n * HID + k] : 0.f);
    }
}

// --------------------------- edge scatter (sum) -----------------------------
// Layer 1: one thread = one (edge, 4-float chunk) of x, 128 feats -> 32 chunks
__global__ void scatter1_kernel(const long long* __restrict__ ei,
                                const float* __restrict__ x,
                                float* __restrict__ agg) {
    int idx = blockIdx.x * blockDim.x + threadIdx.x;
    int e = idx >> 5;
    if (e >= EE) return;
    int c = (idx & 31) * 4;
    size_t src = (size_t)ei[e];
    size_t dst = (size_t)ei[EE + e];
    float4 v = *(const float4*)(x + src * F_IN + c);
    float* a = agg + dst * F_IN + c;
    atomicAdd(a + 0, v.x);
    atomicAdd(a + 1, v.y);
    atomicAdd(a + 2, v.z);
    atomicAdd(a + 3, v.w);
}

// Layer 2: one thread = one (edge, 4-half chunk) of h1, 256 feats -> 64 chunks
typedef __attribute__((ext_vector_type(4))) _Float16 v4h;
__global__ void scatter2_kernel(const long long* __restrict__ ei,
                                const _Float16* __restrict__ h1,
                                float* __restrict__ agg) {
    int idx = blockIdx.x * blockDim.x + threadIdx.x;
    int e = idx >> 6;
    if (e >= EE) return;
    int c = (idx & 63) * 4;
    size_t src = (size_t)ei[e];
    size_t dst = (size_t)ei[EE + e];
    v4h v = *(const v4h*)(h1 + src * HID + c);
    float* a = agg + dst * HID + c;
    atomicAdd(a + 0, (float)v[0]);
    atomicAdd(a + 1, (float)v[1]);
    atomicAdd(a + 2, (float)v[2]);
    atomicAdd(a + 3, (float)v[3]);
}

// ------------------------------ layer 1 GEMM --------------------------------
// h1[m][n] = relu( (agg[m]*inv_deg[m]) @ W1L[n]  +  x[m] @ W1R[n]  + b1[n] )
// Block = 128 threads (4 waves) = one 16-row M-tile; each wave owns 4 N-tiles.
__global__ __launch_bounds__(128) void gemm1_kernel(
        const float* __restrict__ x, const float* __restrict__ agg,
        const float* __restrict__ invdeg,
        const _Float16* __restrict__ W1L, const _Float16* __restrict__ W1R,
        const float* __restrict__ b1, _Float16* __restrict__ h1) {
    const int mtile = blockIdx.x;
    const int lane  = threadIdx.x & 31;
    const int wave  = threadIdx.x >> 5;
    const int half  = lane >> 4;      // lane-half selects K sub-stripe
    const int lm    = lane & 15;
    const int m     = mtile * 16 + lm;

    const float sc = invdeg[m];
    const float* __restrict__ xrow = x   + (size_t)m * F_IN;
    const float* __restrict__ arow = agg + (size_t)m * F_IN;

    // Build all A fragments once (K = 128 -> 4 steps of 32), reuse over N-tiles.
    v16h Ax[4], Ag[4];
#pragma unroll
    for (int t = 0; t < 4; ++t) {
        int k0 = t * 32 + half * 8;
        Ax[t] = load_a_f32(xrow, k0, 1.0f);
        Ag[t] = load_a_f32(arow, k0, sc);
    }

#pragma unroll
    for (int j = 0; j < 4; ++j) {
        const int nb = (wave * 4 + j) * 16;
        const int n  = nb + lm;
        const _Float16* __restrict__ bl = W1L + (size_t)n * F_IN + half * 8;
        const _Float16* __restrict__ br = W1R + (size_t)n * F_IN + half * 8;
        v8f acc = {};
#pragma unroll
        for (int t = 0; t < 4; ++t) {
            v16h Bl = load_a_f16(bl, t * 32);
            acc = __builtin_amdgcn_wmma_f32_16x16x32_f16(false, Ag[t], false, Bl,
                                                         (short)0, acc, false, false);
            v16h Br = load_a_f16(br, t * 32);
            acc = __builtin_amdgcn_wmma_f32_16x16x32_f16(false, Ax[t], false, Br,
                                                         (short)0, acc, false, false);
        }
        const float bias = b1[n];
        const int mrow = mtile * 16 + half * 8;
#pragma unroll
        for (int v = 0; v < 8; ++v) {
            float r = acc[v] + bias;
            r = r > 0.f ? r : 0.f;                 // ReLU
            h1[(size_t)(mrow + v) * HID + n] = (_Float16)r;
        }
    }
}

// ------------------------ layer 2 GEMM + log_softmax ------------------------
// out[m][c] = logsoftmax_c( (agg2[m]*inv_deg[m]) @ W2L + h1[m] @ W2R + b2 )
// W2 padded to 16 columns; columns 4..15 are zero. One wave per M-tile.
__global__ __launch_bounds__(128) void gemm2_kernel(
        const _Float16* __restrict__ h1, const float* __restrict__ agg2,
        const float* __restrict__ invdeg,
        const _Float16* __restrict__ W2L, const _Float16* __restrict__ W2R,
        const float* __restrict__ b2, float* __restrict__ out) {
    const int mtile = blockIdx.x * 4 + (threadIdx.x >> 5);
    if (mtile >= MTILES) return;       // whole-wave exit: EXEC all-ones at WMMA
    const int lane = threadIdx.x & 31;
    const int half = lane >> 4;
    const int lm   = lane & 15;
    const int m    = mtile * 16 + lm;

    const float sc = invdeg[m];
    const float* __restrict__ arow = agg2 + (size_t)m * HID;
    const _Float16* __restrict__ hrow = h1 + (size_t)m * HID;
    const _Float16* __restrict__ bl = W2L + (size_t)lm * HID + half * 8;
    const _Float16* __restrict__ br = W2R + (size_t)lm * HID + half * 8;

    v8f acc = {};
#pragma unroll
    for (int t = 0; t < 8; ++t) {      // K = 256 -> 8 steps of 32
        int k0 = t * 32 + half * 8;
        v16h Aa = load_a_f32(arow, k0, sc);
        v16h Bl = load_a_f16(bl, t * 32);
        acc = __builtin_amdgcn_wmma_f32_16x16x32_f16(false, Aa, false, Bl,
                                                     (short)0, acc, false, false);
        v16h Ah = load_a_f16(hrow, k0);
        v16h Br = load_a_f16(br, t * 32);
        acc = __builtin_amdgcn_wmma_f32_16x16x32_f16(false, Ah, false, Br,
                                                     (short)0, acc, false, false);
    }

    const float bias = (lm < 4) ? b2[lm] : 0.f;
    const int mrow = mtile * 16 + half * 8;
#pragma unroll
    for (int v = 0; v < 8; ++v) {
        float t = acc[v] + bias;
        // log-softmax across the 4 class lanes (groups {0..3},{16..19} carry data)
        float mx = fmaxf(t, __shfl_xor(t, 1));
        mx = fmaxf(mx, __shfl_xor(mx, 2));
        float e = expf(t - mx);
        float s = e + __shfl_xor(e, 1);
        s += __shfl_xor(s, 2);
        float r = (t - mx) - logf(s);
        if (lm < 4) out[(size_t)(mrow + v) * 4 + lm] = r;
    }
}

// ------------------------------- launcher -----------------------------------
extern "C" void kernel_launch(void* const* d_in, const int* in_sizes, int n_in,
                              void* d_out, int out_size, void* d_ws, size_t ws_size,
                              hipStream_t stream) {
    const float*     x   = (const float*)d_in[0];
    const long long* ei  = (const long long*)d_in[1];
    const float*     w1l = (const float*)d_in[2];
    const float*     w1r = (const float*)d_in[3];
    const float*     b1  = (const float*)d_in[4];
    const float*     w2l = (const float*)d_in[5];
    const float*     w2r = (const float*)d_in[6];
    const float*     b2  = (const float*)d_in[7];
    float*           out = (float*)d_out;

    // workspace carve-up (256B aligned slabs)
    char* p = (char*)d_ws;
    auto carve = [&](size_t bytes) { char* r = p; p += (bytes + 255) & ~(size_t)255; return r; };
    float*     deg = (float*)    carve((size_t)NN * 4);            //  0.4 MB (deg -> inv_deg)
    float*     agg = (float*)    carve((size_t)NN * HID * 4);      // 102.4 MB (shared both layers)
    _Float16*  h1  = (_Float16*) carve((size_t)NN * HID * 2);      //  51.2 MB
    _Float16*  W1L = (_Float16*) carve((size_t)HID * F_IN * 2);
    _Float16*  W1R = (_Float16*) carve((size_t)HID * F_IN * 2);
    _Float16*  W2L = (_Float16*) carve((size_t)16 * HID * 2);
    _Float16*  W2R = (_Float16*) carve((size_t)16 * HID * 2);

    const int B = 256;

    // weights -> f16 (and pad W2 to 16 columns)
    prep_weights_kernel<<<(HID * F_IN + B - 1) / B, B, 0, stream>>>(
        w1l, w1r, w2l, w2r, W1L, W1R, W2L, W2R);

    // degrees
    zero_f4_kernel<<<(NN / 4 + B - 1) / B, B, 0, stream>>>((float4*)deg, NN / 4);
    deg_kernel<<<(EE + B - 1) / B, B, 0, stream>>>(ei, deg);
    invdeg_kernel<<<(NN + B - 1) / B, B, 0, stream>>>(deg);

    // layer 1: scatter-sum x then WMMA GEMM (+bias, ReLU) -> h1 (f16)
    {
        int n4 = NN * F_IN / 4;
        zero_f4_kernel<<<(n4 + B - 1) / B, B, 0, stream>>>((float4*)agg, n4);
        long long thr = (long long)EE * 32;
        scatter1_kernel<<<(unsigned)((thr + B - 1) / B), B, 0, stream>>>(ei, x, agg);
        gemm1_kernel<<<MTILES, 128, 0, stream>>>(x, agg, deg, W1L, W1R, b1, h1);
    }

    // layer 2: scatter-sum h1 then WMMA GEMM (+bias, log_softmax) -> out
    {
        int n4 = NN * HID / 4;
        zero_f4_kernel<<<(n4 + B - 1) / B, B, 0, stream>>>((float4*)agg, n4);
        long long thr = (long long)EE * 64;
        scatter2_kernel<<<(unsigned)((thr + B - 1) / B), B, 0, stream>>>(ei, h1, agg);
        gemm2_kernel<<<(MTILES + 3) / 4, 128, 0, stream>>>(h1, agg, deg, W2L, W2R, b2, out);
    }
}
